// OrganicGrowthTransformer_77438260347451
// MI455X (gfx1250) — compile-verified, matrix operand order checked by reading, not run on hardware
//
#include <hip/hip_runtime.h>
#include <hip/hip_bf16.h>
#include <math.h>

// ---------------- model dims (fixed by reference) ----------------
#define VOCAB 50257
#define BSZ   4
#define SEQ   2048
#define DEMB  256
#define HDIM  512
#define NHEAD 8
#define NBLK  2
#define NDOM  8
#define EDIM  32
#define PDIM  5
#define HD    64   // HDIM/NHEAD

typedef __attribute__((ext_vector_type(16))) __bf16 v16bf;
typedef __attribute__((ext_vector_type(8)))  float  v8f;

__device__ __forceinline__ v8f v8f_zero() {
    v8f z;
#pragma unroll
    for (int i = 0; i < 8; ++i) z[i] = 0.f;
    return z;
}

// Load one 16x32 bf16 A/B^T-style fragment. p points at (row*ld + k0 + hh*8):
// vector elems 0..7 <- p[0..7], elems 8..15 <- p[16..23]  (ISA 7.12.2 16-bit A layout)
__device__ __forceinline__ v16bf frag_ld(const __bf16* p) {
    union { v16bf v; uint4 q[2]; } u;
    u.q[0] = *reinterpret_cast<const uint4*>(p);
    u.q[1] = *reinterpret_cast<const uint4*>(p + 16);
    return u.v;
}

__device__ __forceinline__ v8f wmma_bf16(v16bf a, v16bf b, v8f c) {
    return __builtin_amdgcn_wmma_f32_16x16x32_bf16(false, a, false, b, (short)0, c, false, false);
}

// async 16B global -> LDS copy (CDNA5 GLOBAL_LOAD_ASYNC_TO_LDS_B128, ASYNCcnt)
__device__ __forceinline__ void async_cp16(unsigned int lds_off, const void* g) {
    asm volatile("global_load_async_to_lds_b128 %0, %1, off"
                 :: "v"(lds_off), "v"(g) : "memory");
}

// ---------------- f32 -> bf16 convert ----------------
__global__ __launch_bounds__(256) void k_cvt_bf16(const float* __restrict__ in,
                                                  __bf16* __restrict__ out, int n) {
    int i = blockIdx.x * 256 + threadIdx.x;
    if (i < n) out[i] = (__bf16)in[i];
}

// ---------------- embedding + domain correction + pos-enc -> bf16 ----------------
__global__ __launch_bounds__(256) void k_embed(const float* __restrict__ base_emb,
                                               const float* __restrict__ dom_exp,
                                               const float* __restrict__ dom_proj,
                                               const int*   __restrict__ x,
                                               const int*   __restrict__ tdom,
                                               __bf16* __restrict__ h0) {
    __shared__ float se[DEMB];
    __shared__ float sx[EDIM];
    const int t   = blockIdx.x;           // token index in [0, B*S)
    const int k   = threadIdx.x;          // dim in [0, 256)
    const int tok = x[t];
    const int d   = tdom[tok];
    se[k] = base_emb[(size_t)tok * DEMB + k];
    __syncthreads();
    float corr = 0.f;
    if (d >= 0) {
        if (k < EDIM) {
            const float* w = dom_exp + ((size_t)d * EDIM + k) * DEMB;
            float a = 0.f;
            for (int i = 0; i < DEMB; ++i) a += w[i] * se[i];
            sx[k] = a;
        }
        __syncthreads();
        const float* wp = dom_proj + ((size_t)d * DEMB + k) * EDIM;
#pragma unroll
        for (int j = 0; j < EDIM; ++j) corr += sx[j] * wp[j];
    }
    const int s = t & (SEQ - 1);
    float ang = (float)s * __expf(-logf(10000.f) * (float)(k & ~1) / (float)DEMB);
    float pe  = (k & 1) ? __cosf(ang) : __sinf(ang);
    h0[(size_t)t * DEMB + k] = (__bf16)(se[k] + 0.1f * corr + pe);
}

// ---------------- generic WMMA GEMM: C[M,N] = A[M,K] * W[N,K]^T + bias ----------------
// 256 thr = 8 waves (4x2), wave tile 32x64, block tile 128x128.
// A/W 128x32 K-step tiles staged via async global->LDS (double buffered),
// padded LDS rows (40 elems = 80B) for conflict-free ds_load_b128 fragments.
#define GF_F32  1
#define GF_BF16 2
#define GF_RELU 4
#define LROW    40

__global__ __launch_bounds__(256) void k_gemm(const __bf16* __restrict__ A,
                                              const __bf16* __restrict__ W,
                                              const float*  __restrict__ bias,
                                              float*  __restrict__ O32,
                                              __bf16* __restrict__ O16,
                                              int M, int N, int K, int flags) {
    __shared__ __bf16 ldsA[2][128 * LROW];
    __shared__ __bf16 ldsW[2][128 * LROW];
    const int tid  = threadIdx.x;
    const int lane = tid & 31;
    const int wave = tid >> 5;
    const int wm   = wave >> 1, wn = wave & 1;       // 4 x 2 wave grid
    const int rowb = blockIdx.x * 128;
    const int colb = blockIdx.y * 128;
    const int hh   = lane >> 4;
    const int r16  = lane & 15;

    // staging: 512 x 16B chunks per 128x32 tile -> 2 chunks per thread per tile
    const int cr0 = tid >> 2,          cs0 = tid & 3;
    const int cr1 = (tid + 256) >> 2,  cs1 = (tid + 256) & 3;

    v8f acc[2][4];
#pragma unroll
    for (int i = 0; i < 2; ++i)
#pragma unroll
        for (int j = 0; j < 4; ++j) acc[i][j] = v8f_zero();

    auto issue = [&](int k0, int buf) {
        async_cp16((unsigned int)(uintptr_t)&ldsA[buf][cr0 * LROW + cs0 * 8],
                   A + (size_t)(rowb + cr0) * K + k0 + cs0 * 8);
        async_cp16((unsigned int)(uintptr_t)&ldsA[buf][cr1 * LROW + cs1 * 8],
                   A + (size_t)(rowb + cr1) * K + k0 + cs1 * 8);
        async_cp16((unsigned int)(uintptr_t)&ldsW[buf][cr0 * LROW + cs0 * 8],
                   W + (size_t)(colb + cr0) * K + k0 + cs0 * 8);
        async_cp16((unsigned int)(uintptr_t)&ldsW[buf][cr1 * LROW + cs1 * 8],
                   W + (size_t)(colb + cr1) * K + k0 + cs1 * 8);
    };

    const int nsteps = K >> 5;
    issue(0, 0);
    for (int i = 0; i < nsteps; ++i) {
        const int buf = i & 1;
        if (i + 1 < nsteps) {
            // pre-stage HBM->L2 one further tile ahead, then refill other buffer
            __builtin_prefetch(A + (size_t)(rowb + cr0) * K + ((i + 2) << 5), 0, 1);
            __builtin_prefetch(W + (size_t)(colb + cr0) * K + ((i + 2) << 5), 0, 1);
            issue((i + 1) << 5, buf ^ 1);
            asm volatile("s_wait_asynccnt 0x4" ::: "memory");   // this step's 4 copies done
        } else {
            asm volatile("s_wait_asynccnt 0x0" ::: "memory");
        }
        __syncthreads();                                        // all waves' tiles visible

        const __bf16* la = &ldsA[buf][(wm * 32 + r16) * LROW + hh * 8];
        const __bf16* lw = &ldsW[buf][(wn * 64 + r16) * LROW + hh * 8];
        v16bf a0 = frag_ld(la);
        v16bf a1 = frag_ld(la + 16 * LROW);
        v16bf b0 = frag_ld(lw);
        v16bf b1 = frag_ld(lw + 16 * LROW);
        v16bf b2 = frag_ld(lw + 32 * LROW);
        v16bf b3 = frag_ld(lw + 48 * LROW);
        acc[0][0] = wmma_bf16(a0, b0, acc[0][0]);
        acc[0][1] = wmma_bf16(a0, b1, acc[0][1]);
        acc[0][2] = wmma_bf16(a0, b2, acc[0][2]);
        acc[0][3] = wmma_bf16(a0, b3, acc[0][3]);
        acc[1][0] = wmma_bf16(a1, b0, acc[1][0]);
        acc[1][1] = wmma_bf16(a1, b1, acc[1][1]);
        acc[1][2] = wmma_bf16(a1, b2, acc[1][2]);
        acc[1][3] = wmma_bf16(a1, b3, acc[1][3]);
        __syncthreads();                                        // done reading before refill
    }

#pragma unroll
    for (int mt = 0; mt < 2; ++mt)
#pragma unroll
        for (int nt = 0; nt < 4; ++nt)
#pragma unroll
            for (int r = 0; r < 8; ++r) {
                int row = rowb + wm * 32 + mt * 16 + r + 8 * hh;
                int col = colb + wn * 64 + nt * 16 + r16;
                float v = acc[mt][nt][r] + bias[col];
                if (flags & GF_RELU) v = fmaxf(v, 0.f);
                size_t o = (size_t)row * N + col;
                if (flags & GF_F32)  O32[o] = v;
                if (flags & GF_BF16) O16[o] = (__bf16)v;
            }
}

// ---------------- qkv split: q scaled 1/sqrt(HD), v stored transposed ----------------
__global__ __launch_bounds__(256) void k_split_qkv(const __bf16* __restrict__ qkv,
                                                   __bf16* __restrict__ Q,
                                                   __bf16* __restrict__ Km,
                                                   __bf16* __restrict__ Vt) {
    int idx = blockIdx.x * 256 + threadIdx.x;          // over B*S*H
    if (idx >= BSZ * SEQ * HDIM) return;
    int c = idx & (HDIM - 1);
    int s = (idx >> 9) & (SEQ - 1);
    int b = idx >> 20;
    int h = c >> 6, d = c & (HD - 1);
    int bh = b * NHEAD + h;
    const __bf16* row = qkv + ((size_t)b * SEQ + s) * (3 * HDIM);
    float qv = (float)row[c] * 0.125f;                 // 1/sqrt(64)
    Q [((size_t)bh * SEQ + s) * HD + d] = (__bf16)qv;
    Km[((size_t)bh * SEQ + s) * HD + d] = row[HDIM + c];
    Vt[((size_t)bh * HD + d) * SEQ + s] = row[2 * HDIM + c];
}

// ---------------- flash attention (non-causal), 1 wave = 16 query rows ----------------
__global__ __launch_bounds__(128) void k_flash(const __bf16* __restrict__ Q,
                                               const __bf16* __restrict__ Km,
                                               const __bf16* __restrict__ Vt,
                                               __bf16* __restrict__ O) {
    __shared__ __bf16 ldsP[4][16][32];
    const int lane = threadIdx.x & 31;
    const int wave = threadIdx.x >> 5;
    const int hh = lane >> 4, r16 = lane & 15;
    const int b = blockIdx.z, h = blockIdx.y;
    const int bh = b * NHEAD + h;
    const int q0 = blockIdx.x * 64 + wave * 16;

    const __bf16* qp = Q + ((size_t)bh * SEQ + q0 + r16) * HD + hh * 8;
    v16bf qa0 = frag_ld(qp);
    v16bf qa1 = frag_ld(qp + 32);

    float m[8], l[8];
    v8f acc[4];
#pragma unroll
    for (int r = 0; r < 8; ++r) { m[r] = -INFINITY; l[r] = 0.f; }
#pragma unroll
    for (int n = 0; n < 4; ++n) acc[n] = v8f_zero();

    for (int s0 = 0; s0 < SEQ; s0 += 32) {
        v8f st[2];
#pragma unroll
        for (int j = 0; j < 2; ++j) {
            const __bf16* kp = Km + ((size_t)bh * SEQ + s0 + j * 16 + r16) * HD + hh * 8;
            v16bf kb0 = frag_ld(kp);
            v16bf kb1 = frag_ld(kp + 32);
            v8f s = v8f_zero();
            s = wmma_bf16(qa0, kb0, s);
            s = wmma_bf16(qa1, kb1, s);
            st[j] = s;
        }
        float al[8];
#pragma unroll
        for (int r = 0; r < 8; ++r) {
            float rm = fmaxf(st[0][r], st[1][r]);        // row chunk max
            rm = fmaxf(rm, __shfl_xor(rm, 1));
            rm = fmaxf(rm, __shfl_xor(rm, 2));
            rm = fmaxf(rm, __shfl_xor(rm, 4));
            rm = fmaxf(rm, __shfl_xor(rm, 8));
            float mn = fmaxf(m[r], rm);
            al[r] = __expf(m[r] - mn);
            float p0 = __expf(st[0][r] - mn);
            float p1 = __expf(st[1][r] - mn);
            st[0][r] = p0; st[1][r] = p1;
            float rs = p0 + p1;
            rs += __shfl_xor(rs, 1);
            rs += __shfl_xor(rs, 2);
            rs += __shfl_xor(rs, 4);
            rs += __shfl_xor(rs, 8);
            l[r] = l[r] * al[r] + rs;
            m[r] = mn;
        }
#pragma unroll
        for (int n = 0; n < 4; ++n)
#pragma unroll
            for (int r = 0; r < 8; ++r) acc[n][r] *= al[r];
        // C-layout f32 -> A-layout bf16 via LDS (same-wave dep, DScnt handled by compiler)
#pragma unroll
        for (int j = 0; j < 2; ++j)
#pragma unroll
            for (int r = 0; r < 8; ++r)
                ldsP[wave][r + 8 * hh][j * 16 + r16] = (__bf16)st[j][r];
        v16bf pa = frag_ld(&ldsP[wave][r16][hh * 8]);
#pragma unroll
        for (int n = 0; n < 4; ++n) {
            const __bf16* vp = Vt + ((size_t)bh * HD + n * 16 + r16) * SEQ + s0 + hh * 8;
            v16bf vb = frag_ld(vp);
            acc[n] = wmma_bf16(pa, vb, acc[n]);
        }
    }
#pragma unroll
    for (int n = 0; n < 4; ++n)
#pragma unroll
        for (int r = 0; r < 8; ++r) {
            int row = q0 + r + 8 * hh;
            int col = h * HD + n * 16 + r16;
            O[((size_t)b * SEQ + row) * HDIM + col] = (__bf16)(acc[n][r] / l[r]);
        }
}

// ---------------- layernorm( x + delta ) -> f32 + bf16 ----------------
__global__ __launch_bounds__(256) void k_ln(const float* __restrict__ X,
                                            const float* __restrict__ Dl,
                                            const float* __restrict__ g,
                                            const float* __restrict__ bta,
                                            float* __restrict__ Y,
                                            __bf16* __restrict__ Ybf) {
    __shared__ float red[256];
    const int row = blockIdx.x, t = threadIdx.x;
    const float* xr = X  + (size_t)row * HDIM;
    const float* dr = Dl + (size_t)row * HDIM;
    float v0 = xr[t] + dr[t];
    float v1 = xr[t + 256] + dr[t + 256];
    red[t] = v0 + v1; __syncthreads();
    for (int o = 128; o > 0; o >>= 1) { if (t < o) red[t] += red[t + o]; __syncthreads(); }
    float mu = red[0] * (1.f / HDIM); __syncthreads();
    float d0 = v0 - mu, d1 = v1 - mu;
    red[t] = d0 * d0 + d1 * d1; __syncthreads();
    for (int o = 128; o > 0; o >>= 1) { if (t < o) red[t] += red[t + o]; __syncthreads(); }
    float rs = rsqrtf(red[0] * (1.f / HDIM) + 1e-5f);
    float y0 = d0 * rs * g[t] + bta[t];
    float y1 = d1 * rs * g[t + 256] + bta[t + 256];
    size_t o0 = (size_t)row * HDIM + t;
    Y[o0] = y0; Y[o0 + 256] = y1;
    Ybf[o0] = (__bf16)y0; Ybf[o0 + 256] = (__bf16)y1;
}

// ---------------- final LN of last token only: hl = LN(h[:, -1] + infl; onorm) ----------------
__global__ __launch_bounds__(256) void k_ln_last(const float* __restrict__ Hf,
                                                 const float* __restrict__ infl,
                                                 const float* __restrict__ g,
                                                 const float* __restrict__ bta,
                                                 float* __restrict__ hl) {
    __shared__ float red[256];
    const int b = blockIdx.x, t = threadIdx.x;
    const float* xr = Hf + ((size_t)b * SEQ + (SEQ - 1)) * HDIM;
    float v0 = xr[t] + infl[b * HDIM + t];
    float v1 = xr[t + 256] + infl[b * HDIM + t + 256];
    red[t] = v0 + v1; __syncthreads();
    for (int o = 128; o > 0; o >>= 1) { if (t < o) red[t] += red[t + o]; __syncthreads(); }
    float mu = red[0] * (1.f / HDIM); __syncthreads();
    float d0 = v0 - mu, d1 = v1 - mu;
    red[t] = d0 * d0 + d1 * d1; __syncthreads();
    for (int o = 128; o > 0; o >>= 1) { if (t < o) red[t] += red[t + o]; __syncthreads(); }
    float rs = rsqrtf(red[0] * (1.f / HDIM) + 1e-5f);
    hl[b * HDIM + t]       = d0 * rs * g[t] + bta[t];
    hl[b * HDIM + t + 256] = d1 * rs * g[t + 256] + bta[t + 256];
}

// ---------------- tiny f32 matvec: out[m,n] = act(A[m,:]·W[n,:] + bias[n]) ----------------
__global__ __launch_bounds__(256) void k_smv(const float* __restrict__ A, int lda,
                                             const float* __restrict__ W,
                                             const float* __restrict__ bias,
                                             float* __restrict__ out,
                                             int M, int N, int K, int act) {
    int idx = blockIdx.x * 256 + threadIdx.x;
    if (idx >= M * N) return;
    int mrow = idx / N, n = idx - mrow * N;
    const float* a = A + (size_t)mrow * lda;
    const float* w = W + (size_t)n * K;
    float acc = bias[n];
    for (int k = 0; k < K; ++k) acc += a[k] * w[k];
    if (act == 1) acc = 0.5f * acc * (1.f + erff(acc * 0.70710678118654752f));   // exact gelu
    else if (act == 2) acc = 1.f / (1.f + __expf(-acc));                          // sigmoid
    out[idx] = acc;
}

// ---------------- argmax over PD scenarios, copy best scenario row ----------------
__global__ __launch_bounds__(128) void k_pick_best(const float* __restrict__ qual,
                                                   const float* __restrict__ scen,
                                                   float* __restrict__ bsc) {
    __shared__ int sbest;
    const int b = blockIdx.x, t = threadIdx.x;
    if (t == 0) {
        int best = 0; float bq = qual[b * PDIM];
        for (int j = 1; j < PDIM; ++j) {
            float q = qual[b * PDIM + j];
            if (q > bq) { bq = q; best = j; }   // strict > : first max (matches jnp.argmax)
        }
        sbest = best;
    }
    __syncthreads();
    for (int j = t; j < HDIM; j += 128)
        bsc[b * HDIM + j] = scen[((size_t)b * PDIM + sbest) * HDIM + j];
}

// ---------------- fast head helpers ----------------
__device__ __forceinline__ float dotn(const float* __restrict__ a,
                                      const float* __restrict__ b, int n) {
    float s = 0.f;
    for (int k = 0; k < n; ++k) s += a[k] * b[k];
    return s;
}

__global__ __launch_bounds__(256) void k_fast(const int* __restrict__ x,
                                              const float* __restrict__ sw,
                                              const float* __restrict__ se0, const float* __restrict__ se1,
                                              const float* __restrict__ se2, const float* __restrict__ se3,
                                              const float* __restrict__ so0, const float* __restrict__ so1,
                                              const float* __restrict__ so2, const float* __restrict__ so3,
                                              const float* __restrict__ sb0, const float* __restrict__ sb1,
                                              const float* __restrict__ sb2, const float* __restrict__ sb3,
                                              float* __restrict__ out) {
    int idx = blockIdx.x * 256 + threadIdx.x;
    if (idx >= BSZ * VOCAB) return;
    int b = idx / VOCAB, v = idx - b * VOCAB;
    float w0 = sw[0], w1 = sw[1], w2 = sw[2], w3 = sw[3];
    float mw = fmaxf(fmaxf(w0, w1), fmaxf(w2, w3));
    float e0 = __expf(w0 - mw), e1 = __expf(w1 - mw), e2 = __expf(w2 - mw), e3 = __expf(w3 - mw);
    float inv = 1.f / (e0 + e1 + e2 + e3);
    int last = x[b * SEQ + (SEQ - 1)];
    float acc = 0.f;
    acc += e0 * inv * (dotn(se0 + (size_t)last * 16,  so0 + (size_t)v * 16,  16)  + sb0[v]);
    acc += e1 * inv * (dotn(se1 + (size_t)last * 32,  so1 + (size_t)v * 32,  32)  + sb1[v]);
    acc += e2 * inv * (dotn(se2 + (size_t)last * 64,  so2 + (size_t)v * 64,  64)  + sb2[v]);
    acc += e3 * inv * (dotn(se3 + (size_t)last * 128, so3 + (size_t)v * 128, 128) + sb3[v]);
    out[idx] = acc;
}

// ---------------- slow head: out += hl[b] · slow_w[v] + slow_b[v] ----------------
__global__ __launch_bounds__(256) void k_slow(const float* __restrict__ hl,
                                              const float* __restrict__ sw,
                                              const float* __restrict__ sb,
                                              float* __restrict__ out) {
    int idx = blockIdx.x * 256 + threadIdx.x;
    if (idx >= BSZ * VOCAB) return;
    int b = idx / VOCAB, v = idx - b * VOCAB;
    out[idx] += dotn(hl + (size_t)b * HDIM, sw + (size_t)v * HDIM, HDIM) + sb[v];
}

// ================================================================
extern "C" void kernel_launch(void* const* d_in, const int* in_sizes, int n_in,
                              void* d_out, int out_size, void* d_ws, size_t ws_size,
                              hipStream_t stream) {
    (void)in_sizes; (void)n_in; (void)out_size; (void)ws_size;
    const float* base_emb = (const float*)d_in[0];
    const float* dom_exp  = (const float*)d_in[1];
    const float* dom_proj = (const float*)d_in[2];
    const float* se[4]  = {(const float*)d_in[3],  (const float*)d_in[4],
                           (const float*)d_in[5],  (const float*)d_in[6]};
    const float* sow[4] = {(const float*)d_in[7],  (const float*)d_in[8],
                           (const float*)d_in[9],  (const float*)d_in[10]};
    const float* sob[4] = {(const float*)d_in[11], (const float*)d_in[12],
                           (const float*)d_in[13], (const float*)d_in[14]};
    const float* speed_w   = (const float*)d_in[15];
    const float* inproj_w  = (const float*)d_in[16];
    const float* inproj_b  = (const float*)d_in[17];
    const float* qkv_w     = (const float*)d_in[18];
    const float* qkv_b     = (const float*)d_in[19];
    const float* ao_w      = (const float*)d_in[20];
    const float* ao_b      = (const float*)d_in[21];
    const float* ln1_g     = (const float*)d_in[22];
    const float* ln1_b     = (const float*)d_in[23];
    const float* ln2_g     = (const float*)d_in[24];
    const float* ln2_b     = (const float*)d_in[25];
    const float* ff1_w     = (const float*)d_in[26];
    const float* ff1_b     = (const float*)d_in[27];
    const float* ff2_w     = (const float*)d_in[28];
    const float* ff2_b     = (const float*)d_in[29];
    const float* gen1_w    = (const float*)d_in[30];
    const float* gen1_b    = (const float*)d_in[31];
    const float* gen2_w    = (const float*)d_in[32];
    const float* gen2_b    = (const float*)d_in[33];
    const float* ev1_w     = (const float*)d_in[34];
    const float* ev1_b     = (const float*)d_in[35];
    const float* ev2_w     = (const float*)d_in[36];
    const float* ev2_b     = (const float*)d_in[37];
    const float* inf_w     = (const float*)d_in[38];
    const float* inf_b     = (const float*)d_in[39];
    const float* onorm_g   = (const float*)d_in[40];
    const float* onorm_b   = (const float*)d_in[41];
    const float* slow_w    = (const float*)d_in[42];
    const float* slow_b    = (const float*)d_in[43];
    const int*   x         = (const int*)d_in[44];
    const int*   tdom      = (const int*)d_in[45];
    float* out = (float*)d_out;

    // ---- workspace carve-up ----
    char* p = (char*)d_ws;
    auto alloc = [&](size_t bytes) { char* r = p; p += (bytes + 255) & ~(size_t)255; return r; };
    const size_t TOK = (size_t)BSZ * SEQ;                 // 8192
    __bf16* w_inproj = (__bf16*)alloc((size_t)HDIM * DEMB * 2);
    __bf16* w_qkv    = (__bf16*)alloc((size_t)NBLK * 3 * HDIM * HDIM * 2);
    __bf16* w_ao     = (__bf16*)alloc((size_t)NBLK * HDIM * HDIM * 2);
    __bf16* w_ff1    = (__bf16*)alloc((size_t)NBLK * 4 * HDIM * HDIM * 2);
    __bf16* w_ff2    = (__bf16*)alloc((size_t)NBLK * HDIM * 4 * HDIM * 2);
    __bf16* h0bf     = (__bf16*)alloc(TOK * DEMB * 2);
    float*  hbuf     = (float*) alloc(TOK * HDIM * 4);
    __bf16* hbf      = (__bf16*)alloc(TOK * HDIM * 2);
    __bf16* qkvbf    = (__bf16*)alloc(TOK * 3 * HDIM * 2);
    __bf16* qbf      = (__bf16*)alloc(TOK * HDIM * 2);
    __bf16* kbf      = (__bf16*)alloc(TOK * HDIM * 2);
    __bf16* vtbf     = (__bf16*)alloc(TOK * HDIM * 2);
    __bf16* obf      = (__bf16*)alloc(TOK * HDIM * 2);
    float*  tmp32    = (float*) alloc(TOK * HDIM * 4);
    __bf16* ffbf     = (__bf16*)alloc(TOK * 4 * HDIM * 2);
    float*  gbuf     = (float*) alloc((size_t)BSZ * 2 * HDIM * 4);
    float*  scen     = (float*) alloc((size_t)BSZ * PDIM * HDIM * 4);
    float*  ebuf     = (float*) alloc((size_t)BSZ * PDIM * HDIM * 4);
    float*  qual     = (float*) alloc((size_t)BSZ * PDIM * 4);
    float*  bsc      = (float*) alloc((size_t)BSZ * HDIM * 4);
    float*  infl     = (float*) alloc((size_t)BSZ * HDIM * 4);
    float*  hl       = (float*) alloc((size_t)BSZ * HDIM * 4);

    auto cvt = [&](const float* src, __bf16* dst, int n) {
        k_cvt_bf16<<<(n + 255) / 256, 256, 0, stream>>>(src, dst, n);
    };
    cvt(inproj_w, w_inproj, HDIM * DEMB);
    cvt(qkv_w,    w_qkv,    NBLK * 3 * HDIM * HDIM);
    cvt(ao_w,     w_ao,     NBLK * HDIM * HDIM);
    cvt(ff1_w,    w_ff1,    NBLK * 4 * HDIM * HDIM);
    cvt(ff2_w,    w_ff2,    NBLK * HDIM * 4 * HDIM);

    // ---- front-end ----
    k_embed<<<(int)TOK, 256, 0, stream>>>(base_emb, dom_exp, dom_proj, x, tdom, h0bf);

    // h = h0 @ inproj_w^T + b   (8192x256 @ 256x512)
    k_gemm<<<dim3(TOK / 128, HDIM / 128), 256, 0, stream>>>(
        h0bf, w_inproj, inproj_b, hbuf, hbf, (int)TOK, HDIM, DEMB, GF_F32 | GF_BF16);

    // ---- transformer blocks ----
    for (int i = 0; i < NBLK; ++i) {
        k_gemm<<<dim3(TOK / 128, (3 * HDIM) / 128), 256, 0, stream>>>(
            hbf, w_qkv + (size_t)i * 3 * HDIM * HDIM, qkv_b + i * 3 * HDIM,
            nullptr, qkvbf, (int)TOK, 3 * HDIM, HDIM, GF_BF16);
        k_split_qkv<<<(int)(TOK * HDIM / 256), 256, 0, stream>>>(qkvbf, qbf, kbf, vtbf);
        k_flash<<<dim3(SEQ / 64, NHEAD, BSZ), 128, 0, stream>>>(qbf, kbf, vtbf, obf);
        k_gemm<<<dim3(TOK / 128, HDIM / 128), 256, 0, stream>>>(
            obf, w_ao + (size_t)i * HDIM * HDIM, ao_b + i * HDIM,
            tmp32, nullptr, (int)TOK, HDIM, HDIM, GF_F32);
        k_ln<<<(int)TOK, 256, 0, stream>>>(hbuf, tmp32, ln1_g + i * HDIM, ln1_b + i * HDIM,
                                           hbuf, hbf);
        k_gemm<<<dim3(TOK / 128, (4 * HDIM) / 128), 256, 0, stream>>>(
            hbf, w_ff1 + (size_t)i * 4 * HDIM * HDIM, ff1_b + i * 4 * HDIM,
            nullptr, ffbf, (int)TOK, 4 * HDIM, HDIM, GF_BF16 | GF_RELU);
        k_gemm<<<dim3(TOK / 128, HDIM / 128), 256, 0, stream>>>(
            ffbf, w_ff2 + (size_t)i * HDIM * 4 * HDIM, ff2_b + i * HDIM,
            tmp32, nullptr, (int)TOK, HDIM, 4 * HDIM, GF_F32);
        k_ln<<<(int)TOK, 256, 0, stream>>>(hbuf, tmp32, ln2_g + i * HDIM, ln2_b + i * HDIM,
                                           hbuf, hbf);
    }

    // ---- generation / evaluation head (B=4, tiny) ----
    k_smv<<<(BSZ * 2 * HDIM + 255) / 256, 256, 0, stream>>>(
        hbuf + (size_t)(SEQ - 1) * HDIM, SEQ * HDIM, gen1_w, gen1_b, gbuf,
        BSZ, 2 * HDIM, HDIM, 1);
    k_smv<<<(BSZ * PDIM * HDIM + 255) / 256, 256, 0, stream>>>(
        gbuf, 2 * HDIM, gen2_w, gen2_b, scen, BSZ, PDIM * HDIM, 2 * HDIM, 0);
    k_smv<<<(BSZ * PDIM * HDIM + 255) / 256, 256, 0, stream>>>(
        scen, HDIM, ev1_w, ev1_b, ebuf, BSZ * PDIM, HDIM, HDIM, 1);
    k_smv<<<1, 256, 0, stream>>>(ebuf, HDIM, ev2_w, ev2_b, qual, BSZ * PDIM, 1, HDIM, 2);
    k_pick_best<<<BSZ, 128, 0, stream>>>(qual, scen, bsc);
    k_smv<<<(BSZ * HDIM + 255) / 256, 256, 0, stream>>>(
        bsc, HDIM, inf_w, inf_b, infl, BSZ, HDIM, HDIM, 0);
    k_ln_last<<<BSZ, 256, 0, stream>>>(hbuf, infl, onorm_g, onorm_b, hl);

    // ---- vocab heads (bandwidth bound, f32) ----
    k_fast<<<(BSZ * VOCAB + 255) / 256, 256, 0, stream>>>(
        x, speed_w, se[0], se[1], se[2], se[3], sow[0], sow[1], sow[2], sow[3],
        sob[0], sob[1], sob[2], sob[3], out);
    k_slow<<<(BSZ * VOCAB + 255) / 256, 256, 0, stream>>>(hl, slow_w, slow_b, out);
}
